// _InteractionLayer_62122406969474
// MI455X (gfx1250) — compile-verified, matrix operand order checked by reading, not run on hardware
//
#include <hip/hip_runtime.h>

typedef __attribute__((ext_vector_type(16))) _Float16 v16h;
typedef __attribute__((ext_vector_type(8)))  _Float16 v8h;
typedef __attribute__((ext_vector_type(8)))  float    v8f;
typedef __attribute__((ext_vector_type(4)))  float    v4f;

#define D_MODEL   256
#define NUM_HEADS 8
#define HEAD_DIM  32
#define SPARSE_K  32

// ---------------------------------------------------------------------------
// Weight pre-pack: W[K,N] f32 row-major  ->  Wt[N,K] f16 (transposed)
// ---------------------------------------------------------------------------
__global__ void wt_pack_kernel(const float* __restrict__ W, _Float16* __restrict__ Wt,
                               int K, int N) {
    int i = blockIdx.x * blockDim.x + threadIdx.x;
    if (i < N * K) {
        int n = i / K, k = i % K;
        Wt[i] = (_Float16)W[(size_t)k * N + n];
    }
}

// ---------------------------------------------------------------------------
// Generic GEMM:  C[M,N] = A[M,K] (f32) * W[K,N] + bias, optional ReLU.
// W supplied pre-packed as Wt[N,K] f16.  One wave -> one 16x16 tile via
// v_wmma_f32_16x16x32_f16; 4 waves/block share an LDS-staged f16 A tile.
// M,N,K all multiples of {16,64,32} respectively in this model.
// ---------------------------------------------------------------------------
__global__ __launch_bounds__(128) void gemm_wmma_kernel(
    const float* __restrict__ A, const _Float16* __restrict__ Wt,
    const float* __restrict__ bias, float* __restrict__ C,
    int M, int N, int K, int relu)
{
    __shared__ _Float16 As[16][32];

    const int wave   = threadIdx.x >> 5;
    const int lane   = threadIdx.x & 31;
    const int m_base = blockIdx.y * 16;
    const int n_base = (blockIdx.x * 4 + wave) * 16;

    // cooperative A staging coords (128 thr load 16x32 f32 -> f16)
    const int arow = threadIdx.x >> 3;       // 0..15
    const int acol = (threadIdx.x & 7) * 4;  // 0..28 step 4

    // fragment coords (16-bit A/B layout: lanes 0-15 K{0..7,16..23}, 16-31 K{8..15,24..31})
    const int fm = lane & 15;
    const int kh = (lane >> 4) * 8;
    const _Float16* wrow = Wt + (size_t)(n_base + fm) * K;

    v8f acc = {};
    for (int kk = 0; kk < K; kk += 32) {
        v4f av = *(const v4f*)(A + (size_t)(m_base + arow) * K + kk + acol);
        As[arow][acol + 0] = (_Float16)av[0];
        As[arow][acol + 1] = (_Float16)av[1];
        As[arow][acol + 2] = (_Float16)av[2];
        As[arow][acol + 3] = (_Float16)av[3];
        if (kk + 32 < K) __builtin_prefetch(wrow + kk + 32, 0, 0);
        __syncthreads();

        v8h alo = *(const v8h*)&As[fm][kh];
        v8h ahi = *(const v8h*)&As[fm][kh + 16];
        v8h blo = *(const v8h*)(wrow + kk + kh);
        v8h bhi = *(const v8h*)(wrow + kk + kh + 16);
        v16h af, bf;
#pragma unroll
        for (int i = 0; i < 8; ++i) {
            af[i] = alo[i]; af[i + 8] = ahi[i];
            bf[i] = blo[i]; bf[i + 8] = bhi[i];
        }
        acc = __builtin_amdgcn_wmma_f32_16x16x32_f16(
                  false, af, false, bf, (short)0, acc, false, false);
        __syncthreads();
    }

    // C layout: VGPR r -> lanes 0-15: (M=r, N=lane); lanes 16-31: (M=8+r, N=lane-16)
    const int nrel = lane & 15;
    const int mhi  = (lane >> 4) * 8;
    const int col  = n_base + nrel;
    const float bv = bias ? bias[col] : 0.0f;
#pragma unroll
    for (int r = 0; r < 8; ++r) {
        float y = acc[r] + bv;
        if (relu) y = y > 0.0f ? y : 0.0f;
        C[(size_t)(m_base + mhi + r) * N + col] = y;
    }
}

// ---------------------------------------------------------------------------
// RoPE in-place on [ntok, 256] with heading == 0.  128 thr/token: (head, pair)
// ---------------------------------------------------------------------------
__global__ __launch_bounds__(128) void rope_kernel(float* __restrict__ x,
                                                   const float* __restrict__ pos)
{
    int row = blockIdx.x;
    int t = threadIdx.x;
    int h = t >> 4, i = t & 15;
    float p    = (i < 8) ? pos[(size_t)row * 2 + 0] : pos[(size_t)row * 2 + 1];
    float invf = powf(10000.0f, -(float)(i & 7) * 0.125f);
    float ang  = p * invf;
    float c = cosf(ang), s = sinf(ang);
    float* px = x + (size_t)row * D_MODEL + h * HEAD_DIM + i * 2;
    float x1 = px[0], x2 = px[1];
    px[0] = x1 * c - x2 * s;
    px[1] = x1 * s + x2 * c;
}

// ---------------------------------------------------------------------------
// Top-K (K=32) nearest keys per query.  Packed (dist_bits<<32)|idx argmin
// extraction gives ascending-distance order with low-index tie break.
// ---------------------------------------------------------------------------
__global__ __launch_bounds__(256) void topk_kernel(
    const float* __restrict__ q_pos, const float* __restrict__ k_pos,
    const unsigned char* __restrict__ k_mask, int* __restrict__ idx_out,
    int Nq, int Nk)
{
    __shared__ unsigned long long skey[1024];
    __shared__ unsigned long long sred[256];
    int row = blockIdx.x;          // b*Nq + q
    int b   = row / Nq;
    int t   = threadIdx.x;
    float qx = q_pos[(size_t)row * 2 + 0];
    float qy = q_pos[(size_t)row * 2 + 1];

    for (int j = t; j < Nk; j += 256) {
        float dx = qx - k_pos[((size_t)b * Nk + j) * 2 + 0];
        float dy = qy - k_pos[((size_t)b * Nk + j) * 2 + 1];
        float d  = dx * dx + dy * dy;
        unsigned int bits = k_mask[(size_t)b * Nk + j] ? __float_as_uint(d) : 0x7f800000u;
        skey[j] = ((unsigned long long)bits << 32) | (unsigned int)j;
    }
    __syncthreads();

    for (int sel = 0; sel < SPARSE_K; ++sel) {
        unsigned long long best = ~0ULL;
        for (int j = t; j < Nk; j += 256) {
            unsigned long long kv = skey[j];
            best = kv < best ? kv : best;
        }
        sred[t] = best;
        __syncthreads();
        for (int off = 128; off > 0; off >>= 1) {
            if (t < off) {
                unsigned long long o = sred[t + off];
                if (o < sred[t]) sred[t] = o;
            }
            __syncthreads();
        }
        if (t == 0) {
            int widx = (int)(sred[0] & 0xffffffffULL);
            idx_out[(size_t)row * SPARSE_K + sel] = widx;
            skey[widx] = ~0ULL;
        }
        __syncthreads();
    }
}

// ---------------------------------------------------------------------------
// Gathered sparse attention core.  1 block per query, 1 wave per head,
// lane = key slot for scores / output dim for the PV accumulation.
// ---------------------------------------------------------------------------
__global__ __launch_bounds__(256) void attn_kernel(
    const float* __restrict__ q, const float* __restrict__ k,
    const float* __restrict__ v, const int* __restrict__ topk,
    const unsigned char* __restrict__ k_mask, float* __restrict__ out,
    int Nq, int Nk)
{
    __shared__ float qs[NUM_HEADS][HEAD_DIM];
    __shared__ float ps[NUM_HEADS][SPARSE_K];
    __shared__ int   kid[NUM_HEADS][SPARSE_K];

    int row  = blockIdx.x;
    int b    = row / Nq;
    int h    = threadIdx.x >> 5;
    int lane = threadIdx.x & 31;

    qs[h][lane] = q[(size_t)row * D_MODEL + h * HEAD_DIM + lane];
    int ki = topk[(size_t)row * SPARSE_K + lane];
    kid[h][lane] = ki;
    __syncthreads();

    const float* kp = k + ((size_t)b * Nk + ki) * D_MODEL + h * HEAD_DIM;
    float s = 0.0f;
#pragma unroll
    for (int d = 0; d < HEAD_DIM; ++d) s += qs[h][d] * kp[d];
    s *= 0.17677669529663687f;                     // 1/sqrt(32)
    if (!k_mask[(size_t)b * Nk + ki]) s = -1.0e9f;

    float m = s;
#pragma unroll
    for (int off = 16; off > 0; off >>= 1) m = fmaxf(m, __shfl_xor(m, off, 32));
    float e = expf(s - m);
    float sum = e;
#pragma unroll
    for (int off = 16; off > 0; off >>= 1) sum += __shfl_xor(sum, off, 32);
    ps[h][lane] = e / sum;
    __syncthreads();

    float acc = 0.0f;
#pragma unroll
    for (int j = 0; j < SPARSE_K; ++j) {
        acc += ps[h][j] * v[((size_t)b * Nk + kid[h][j]) * D_MODEL + h * HEAD_DIM + lane];
    }
    out[(size_t)row * D_MODEL + h * HEAD_DIM + lane] = acc;
}

// ---------------------------------------------------------------------------
// out = LayerNorm(base + add) * g + b, then optional token-mask zeroing.
// out may alias base (each element read before any write; barriers between).
// ---------------------------------------------------------------------------
__global__ __launch_bounds__(256) void resid_ln_kernel(
    const float* __restrict__ base, const float* __restrict__ add,
    const float* __restrict__ g, const float* __restrict__ bshift,
    const unsigned char* __restrict__ mask, float* __restrict__ out)
{
    __shared__ float red[256];
    int row = blockIdx.x, t = threadIdx.x;
    float x = base[(size_t)row * D_MODEL + t] + add[(size_t)row * D_MODEL + t];

    red[t] = x; __syncthreads();
    for (int off = 128; off > 0; off >>= 1) {
        if (t < off) red[t] += red[t + off];
        __syncthreads();
    }
    float mu = red[0] * (1.0f / 256.0f);
    __syncthreads();

    float d = x - mu;
    red[t] = d * d; __syncthreads();
    for (int off = 128; off > 0; off >>= 1) {
        if (t < off) red[t] += red[t + off];
        __syncthreads();
    }
    float var = red[0] * (1.0f / 256.0f);

    float y = d * rsqrtf(var + 1e-5f) * g[t] + bshift[t];
    if (mask && !mask[row]) y = 0.0f;
    out[(size_t)row * D_MODEL + t] = y;
}

// ---------------------------------------------------------------------------
extern "C" void kernel_launch(void* const* d_in, const int* in_sizes, int n_in,
                              void* d_out, int out_size, void* d_ws, size_t ws_size,
                              hipStream_t stream)
{
    (void)in_sizes; (void)n_in; (void)out_size; (void)ws_size;

    const float*         agent_feat = (const float*)d_in[0];
    const float*         map_feat   = (const float*)d_in[1];
    const float*         agent_pos  = (const float*)d_in[2];
    const float*         map_pos    = (const float*)d_in[3];
    const unsigned char* agent_mask = (const unsigned char*)d_in[4];
    const unsigned char* map_mask   = (const unsigned char*)d_in[5];

    // ---- workspace layout (bytes) ----
    char* ws = (char*)d_ws;
    float* region0 = (float*)(ws);                 // 8,388,608 f: q|k|v, later FFN hidden
    float* region1 = (float*)(ws + 33554432);      // 2,097,152 f: attn core out
    float* region2 = (float*)(ws + 41943040);      // 2,097,152 f: projection / FFN out
    int*   idx_mm  = (int*)(ws + 50331648);        // 262,144 i
    int*   idx_aa  = (int*)(ws + 51380224);        // 16,384 i
    int*   idx_am  = (int*)(ws + 51445760);        // 16,384 i
    _Float16* w16  = (_Float16*)(ws + 51511296);   // 2,359,296 halves

    float* qbuf = region0;
    float* kbuf = region0 + 2097152;
    float* vbuf = region0 + 4194304;

    float* outA = (float*)d_out;                   // agent feat [512,256]
    float* outM = (float*)d_out + 131072;          // map feat [8192,256]

    // ---- f16 transposed weight copies ----
    _Float16* mmWq = w16;            _Float16* mmWk = w16 + 65536;
    _Float16* mmWv = w16 + 131072;   _Float16* mmWo = w16 + 196608;
    _Float16* aaWq = w16 + 262144;   _Float16* aaWk = w16 + 327680;
    _Float16* aaWv = w16 + 393216;   _Float16* aaWo = w16 + 458752;
    _Float16* amWq = w16 + 524288;   _Float16* amWk = w16 + 589824;
    _Float16* amWv = w16 + 655360;   _Float16* amWo = w16 + 720896;
    _Float16* mmW1 = w16 + 786432;   _Float16* mmW2 = w16 + 1048576;
    _Float16* aaW1 = w16 + 1310720;  _Float16* aaW2 = w16 + 1572864;
    _Float16* amW1 = w16 + 1835008;  _Float16* amW2 = w16 + 2097152;

    auto pack = [&](int inIdx, _Float16* dst, int K, int N) {
        int total = K * N;
        wt_pack_kernel<<<(total + 255) / 256, 256, 0, stream>>>(
            (const float*)d_in[inIdx], dst, K, N);
    };
    // attn weights: [K=256, N=256]
    pack(6,  mmWq, 256, 256); pack(8,  mmWk, 256, 256);
    pack(10, mmWv, 256, 256); pack(12, mmWo, 256, 256);
    pack(14, aaWq, 256, 256); pack(16, aaWk, 256, 256);
    pack(18, aaWv, 256, 256); pack(20, aaWo, 256, 256);
    pack(22, amWq, 256, 256); pack(24, amWk, 256, 256);
    pack(26, amWv, 256, 256); pack(28, amWo, 256, 256);
    // ffn weights
    pack(34, mmW1, 256, 1024); pack(36, mmW2, 1024, 256);
    pack(42, aaW1, 256, 1024); pack(44, aaW2, 1024, 256);
    pack(50, amW1, 256, 1024); pack(52, amW2, 1024, 256);

    auto gemm = [&](const float* A, const _Float16* Wt, const float* bias,
                    float* C, int M, int N, int K, int relu) {
        dim3 grid(N / 64, M / 16);
        gemm_wmma_kernel<<<grid, 128, 0, stream>>>(A, Wt, bias, C, M, N, K, relu);
    };

    auto run_block = [&](const float* qfeat, const float* kfeat,
                         const float* qp, const float* kp,
                         const unsigned char* kmask, const unsigned char* outmask,
                         int Nq, int Nk, int* idx,
                         const _Float16* Wq, const float* bq,
                         const _Float16* Wk, const float* bk,
                         const _Float16* Wv, const float* bv,
                         const _Float16* Wo, const float* bo,
                         const float* n1g, const float* n1b,
                         const _Float16* W1, const float* b1,
                         const _Float16* W2, const float* b2,
                         const float* n2g, const float* n2b,
                         float* feat_out) {
        const int Mq = 8 * Nq, Mk = 8 * Nk;
        topk_kernel<<<Mq, 256, 0, stream>>>(qp, kp, kmask, idx, Nq, Nk);
        gemm(qfeat, Wq, bq, qbuf, Mq, 256, 256, 0);
        gemm(kfeat, Wk, bk, kbuf, Mk, 256, 256, 0);
        gemm(kfeat, Wv, bv, vbuf, Mk, 256, 256, 0);
        rope_kernel<<<Mq, 128, 0, stream>>>(qbuf, qp);
        rope_kernel<<<Mk, 128, 0, stream>>>(kbuf, kp);
        attn_kernel<<<Mq, 256, 0, stream>>>(qbuf, kbuf, vbuf, idx, kmask, region1, Nq, Nk);
        gemm(region1, Wo, bo, region2, Mq, 256, 256, 0);
        resid_ln_kernel<<<Mq, 256, 0, stream>>>(qfeat, region2, n1g, n1b, nullptr, feat_out);
        gemm(feat_out, W1, b1, region0, Mq, 1024, 256, 1);   // hidden reuses q/k/v region
        gemm(region0, W2, b2, region2, Mq, 256, 1024, 0);
        resid_ln_kernel<<<Mq, 256, 0, stream>>>(feat_out, region2, n2g, n2b, outmask, feat_out);
    };

    // map -> map
    run_block(map_feat, map_feat, map_pos, map_pos, map_mask, map_mask,
              1024, 1024, idx_mm,
              mmWq, (const float*)d_in[7],  mmWk, (const float*)d_in[9],
              mmWv, (const float*)d_in[11], mmWo, (const float*)d_in[13],
              (const float*)d_in[30], (const float*)d_in[31],
              mmW1, (const float*)d_in[35], mmW2, (const float*)d_in[37],
              (const float*)d_in[32], (const float*)d_in[33],
              outM);

    // agent -> agent
    run_block(agent_feat, agent_feat, agent_pos, agent_pos, agent_mask, agent_mask,
              64, 64, idx_aa,
              aaWq, (const float*)d_in[15], aaWk, (const float*)d_in[17],
              aaWv, (const float*)d_in[19], aaWo, (const float*)d_in[21],
              (const float*)d_in[38], (const float*)d_in[39],
              aaW1, (const float*)d_in[43], aaW2, (const float*)d_in[45],
              (const float*)d_in[40], (const float*)d_in[41],
              outA);

    // agent -> map (queries = updated agents, keys/values = updated map)
    run_block(outA, outM, agent_pos, map_pos, map_mask, agent_mask,
              64, 1024, idx_am,
              amWq, (const float*)d_in[23], amWk, (const float*)d_in[25],
              amWv, (const float*)d_in[27], amWo, (const float*)d_in[29],
              (const float*)d_in[46], (const float*)d_in[47],
              amW1, (const float*)d_in[51], amW2, (const float*)d_in[53],
              (const float*)d_in[48], (const float*)d_in[49],
              outA);
}
